// EAGFM_56839597195749
// MI455X (gfx1250) — matrix operand order repaired and checked by reading, not compiled, not run on hardware
//
#include <hip/hip_runtime.h>
#include <hip/hip_bf16.h>

typedef _Float16 half_t;
typedef half_t v16h __attribute__((ext_vector_type(16)));
typedef float  v8f  __attribute__((ext_vector_type(8)));

#define B_ 2
#define C_ 64
#define H_ 256
#define W_ 256
#define HW_ (H_*W_)
#define SLD 264   // padded f16 row stride for the 256x256 score matrix (bank-conflict-free transpose reads)

// ---- CDNA5 async / TDM feature detection (guarded: falls back to plain copies) ----
#if __has_builtin(__builtin_amdgcn_tensor_load_to_lds) && __has_builtin(__builtin_amdgcn_s_wait_tensorcnt)
#define HAVE_TDM 1
#endif
#if __has_builtin(__builtin_amdgcn_global_load_async_to_lds_b128) && __has_builtin(__builtin_amdgcn_s_wait_asynccnt)
#define HAVE_ASYNC 1
#endif

#if defined(HAVE_ASYNC)
// Builtin signature (from hipcc diagnostic): (int4 AS1*, int4 AS3*, imm int, imm int)
typedef int b128i __attribute__((vector_size(16)));
typedef __attribute__((address_space(1))) b128i* gb128_p;
typedef __attribute__((address_space(3))) b128i* lb128_p;
#endif

#if defined(HAVE_TDM)
typedef unsigned int u32x4 __attribute__((ext_vector_type(4)));
typedef int i32x4 __attribute__((ext_vector_type(4)));
typedef int i32x8 __attribute__((ext_vector_type(8)));

// 1-D contiguous TDM copy global -> LDS. nbytes must be a multiple of 8.
// D# built per CDNA5 ISA ch.8: group0 = {count, lds_addr, global_addr, type=2},
// group1 = {data_size=8B, tensor_dim0, tensor_dim1=1, tile_dim0, tile_dim1=1, dim0_stride}.
__device__ __forceinline__ void tdm_load_1d(const void* gsrc, unsigned lds_byte_off, unsigned nbytes) {
  unsigned n8 = nbytes >> 3;                       // elements of 8 bytes
  unsigned long long ga = (unsigned long long)gsrc;
  u32x4 g0;
  g0[0] = 1u;                                      // count=1, user descriptor
  g0[1] = lds_byte_off;                            // lds_addr (bytes)
  g0[2] = (unsigned)ga;                            // global_addr[31:0]
  g0[3] = (unsigned)((ga >> 32) & 0x1ffffffull) | (2u << 30);  // global_addr[56:32] | type=2
  i32x8 g1;
  g1[0] = (int)(3u << 16);                         // workgroup_mask=0 | data_size=3 (8B)
  g1[1] = (int)((n8 & 0xffffu) << 16);             // tensor_dim0[15:0] at bits 63:48
  g1[2] = (int)((n8 >> 16) & 0xffffu) | (1 << 16); // tensor_dim0[31:16] | tensor_dim1=1
  g1[3] = (int)((n8 & 0xffffu) << 16);             // tile_dim0 at bits 127:112
  g1[4] = 1;                                       // tile_dim1=1, tile_dim2=0
  g1[5] = (int)n8;                                 // tensor_dim0_stride[31:0]
  g1[6] = 0;
  g1[7] = 0;
  i32x4 z4 = {0, 0, 0, 0};
#if defined(__clang_major__) && __clang_major__ >= 23
  i32x8 z8 = {0, 0, 0, 0, 0, 0, 0, 0};
  __builtin_amdgcn_tensor_load_to_lds(g0, g1, z4, z4, z8, 0);
#else
  __builtin_amdgcn_tensor_load_to_lds(g0, g1, z4, z4, 0);
#endif
}
#endif

// ---- WMMA helpers: CDNA5 V_WMMA_F32_16X16X32_F16 fragments, contiguous k order (k = 16*half + e) for A and B ----

__device__ __forceinline__ v8f wmma_f16(v16h a, v16h b, v8f c) {
  return __builtin_amdgcn_wmma_f32_16x16x32_f16(false, a, false, b, (short)0, c, false, false);
}

// A element e (of 16 f16) for lane-half hl: K = hl*16 + e; M = lane&15
__device__ __forceinline__ int ka_idx(int e, int hl) {
  return (hl << 4) + e;
}

// A fragment from row-major src[m][k], leading dim ld
__device__ __forceinline__ v16h frag_a_rm(const half_t* src, int ld, int m0, int k0, int lane) {
  int hl = lane >> 4;
  const half_t* p = src + (m0 + (lane & 15)) * ld + k0;
  v16h a;
#pragma unroll
  for (int e = 0; e < 16; ++e) a[e] = p[ka_idx(e, hl)];
  return a;
}

// A fragment where A[m][k] = src[k][m] (transposed source), src row-major with ld
__device__ __forceinline__ v16h frag_a_tr(const half_t* src, int ld, int m0, int k0, int lane) {
  int hl = lane >> 4;
  int m = m0 + (lane & 15);
  v16h a;
#pragma unroll
  for (int e = 0; e < 16; ++e) a[e] = src[(k0 + ka_idx(e, hl)) * ld + m];
  return a;
}

// B fragment from row-major src[k][n]: element e: K = k0 + hl*16 + e; N = n0 + (lane&15)
__device__ __forceinline__ v16h frag_b_rm(const half_t* src, int ld, int k0, int n0, int lane) {
  int hl = lane >> 4;
  int n = n0 + (lane & 15);
  v16h b;
#pragma unroll
  for (int e = 0; e < 16; ++e) b[e] = src[(k0 + (hl << 4) + e) * ld + n];
  return b;
}

// B fragment where B[k][n] = src[n][k] (src row-major [n][k]) -> 16 contiguous f16 per lane
__device__ __forceinline__ v16h frag_b_tr(const half_t* src, int ld, int k0, int n0, int lane) {
  int hl = lane >> 4;
  const half_t* p = src + (n0 + (lane & 15)) * ld + k0 + (hl << 4);
  v16h b;
#pragma unroll
  for (int e = 0; e < 16; ++e) b[e] = p[e];
  return b;
}

// ---- Kernel 1: four 1x1 conv projections as WMMA GEMMs: t = W(64x64) * x(64 x B*HW) + bias ----
__global__ void __launch_bounds__(256) k_conv1x1_qv(
    const float* xl, const float* xr,
    const float* w0, const float* bi0, const float* w1, const float* bi1,
    const float* w2, const float* bi2, const float* w3, const float* bi3,
    half_t* t0, half_t* t1, half_t* t2, half_t* t3) {
  extern __shared__ char smem[];
  half_t* sX = (half_t*)smem;          // 64 x 128 activations (f16)
  half_t* sW = sX + 64 * 128;          // 64 x 64 weights (f16)
  int tid = threadIdx.x, lane = tid & 31, wave = tid >> 5, hl = lane >> 4;

  int which = blockIdx.y;
  const float* x = (which & 1) ? xr : xl;
  const float* w;  const float* bias;  half_t* tout;
  if      (which == 0) { w = w0; bias = bi0; tout = t0; }
  else if (which == 1) { w = w1; bias = bi1; tout = t1; }
  else if (which == 2) { w = w2; bias = bi2; tout = t2; }
  else                 { w = w3; bias = bi3; tout = t3; }

  long n0 = (long)blockIdx.x * 128;          // column over B*HW (HW multiple of 128 -> no batch cross)
  int  b  = (int)(n0 / HW_);
  long hw0 = n0 % HW_;
  const float* xb = x + (size_t)b * C_ * HW_ + hw0;

  for (int i = tid; i < 64 * 128; i += 256) {
    int c = i >> 7, col = i & 127;
    sX[i] = (half_t)xb[(size_t)c * HW_ + col];
  }
  for (int i = tid; i < 64 * 64; i += 256) sW[i] = (half_t)w[i];
  __syncthreads();

  int nt = wave * 16;
  for (int mt = 0; mt < 4; ++mt) {
    v8f acc = {};
#pragma unroll
    for (int kt = 0; kt < 2; ++kt) {
      v16h a = frag_a_rm(sW, 64, mt * 16, kt * 32, lane);
      v16h bb = frag_b_rm(sX, 128, kt * 32, nt, lane);
      acc = wmma_f16(a, bb, acc);
    }
    int col = nt + (lane & 15);
    half_t* outb = tout + (size_t)b * C_ * HW_ + hw0 + col;
#pragma unroll
    for (int v = 0; v < 8; ++v) {
      int o = mt * 16 + hl * 8 + v;
      outb[(size_t)o * HW_] = (half_t)(acc[v] + bias[o]);
    }
  }
}

// ---- Kernel 2: depthwise 3x3 + bias, NCHW f16 -> channels-last f16 [B,H,W,C] ----
__global__ void __launch_bounds__(256) k_dw3x3(
    const half_t* t0, const half_t* t1, const half_t* t2, const half_t* t3,
    const float* w0, const float* bi0, const float* w1, const float* bi1,
    const float* w2, const float* bi2, const float* w3, const float* bi3,
    half_t* Ql, half_t* Qr, half_t* Vl, half_t* Vr) {
  int which = blockIdx.y;
  const half_t* src; const float* w; const float* bias; half_t* dst;
  if      (which == 0) { src = t0; w = w0; bias = bi0; dst = Ql; }
  else if (which == 1) { src = t1; w = w1; bias = bi1; dst = Qr; }
  else if (which == 2) { src = t2; w = w2; bias = bi2; dst = Vl; }
  else                 { src = t3; w = w3; bias = bi3; dst = Vr; }

  long gid = (long)blockIdx.x * blockDim.x + threadIdx.x;
  int x = (int)(gid % W_);  long r = gid / W_;
  int y = (int)(r % H_);    r /= H_;
  int c = (int)(r % C_);
  int b = (int)(r / C_);

  const half_t* plane = src + (size_t)(b * C_ + c) * HW_;
  __builtin_prefetch(plane + (y + 2 < H_ ? (y + 2) : y) * W_ + x, 0, 1);  // global_prefetch_b8
  const float* wk = w + c * 9;
  float acc = bias[c];
#pragma unroll
  for (int dy = -1; dy <= 1; ++dy) {
#pragma unroll
    for (int dx = -1; dx <= 1; ++dx) {
      int yy = y + dy, xx = x + dx;
      if (yy >= 0 && yy < H_ && xx >= 0 && xx < W_)
        acc += wk[(dy + 1) * 3 + (dx + 1)] * (float)plane[yy * W_ + xx];
    }
  }
  dst[(((size_t)b * H_ + y) * W_ + x) * C_ + c] = (half_t)acc;
}

// ---- Kernel 3: full per-row cross attention in LDS (one workgroup per (b,h) row) ----
__global__ void __launch_bounds__(256) k_attn(
    const half_t* Ql, const half_t* Qr, const half_t* Vl, const half_t* Vr,
    half_t* Fr2l, half_t* Fl2r) {
  extern __shared__ char smem[];
  half_t* sQl = (half_t*)smem;            // 256 x 64
  half_t* sQr = sQl + 256 * 64;
  half_t* sVl = sQr + 256 * 64;
  half_t* sVr = sVl + 256 * 64;
  half_t* sS  = sVr + 256 * 64;           // 256 x SLD (f16, padded)
  float* srsum = (float*)(sS + 256 * SLD);
  float* srscl = srsum + 256;
  float* sZ    = srscl + 256;
  float* sred  = sZ + 256;

  int tid = threadIdx.x, lane = tid & 31, wave = tid >> 5, hl = lane >> 4;
  size_t rowbase = (size_t)blockIdx.x * (W_ * C_);

  // stage Q/V tiles (each 256x64 f16 = 32KB)
#if defined(HAVE_TDM)
  if (wave == 0) {
    tdm_load_1d(Ql + rowbase, (unsigned)((char*)sQl - smem), 32768u);
    tdm_load_1d(Qr + rowbase, (unsigned)((char*)sQr - smem), 32768u);
    tdm_load_1d(Vl + rowbase, (unsigned)((char*)sVl - smem), 32768u);
    tdm_load_1d(Vr + rowbase, (unsigned)((char*)sVr - smem), 32768u);
    __builtin_amdgcn_s_wait_tensorcnt(0);
  }
#else
  {
    const uint4* s0 = (const uint4*)(Ql + rowbase);
    const uint4* s1 = (const uint4*)(Qr + rowbase);
    const uint4* s2 = (const uint4*)(Vl + rowbase);
    const uint4* s3 = (const uint4*)(Vr + rowbase);
    uint4* d0 = (uint4*)sQl; uint4* d1 = (uint4*)sQr;
    uint4* d2 = (uint4*)sVl; uint4* d3 = (uint4*)sVr;
    for (int j = tid; j < 2048; j += 256) { d0[j] = s0[j]; d1[j] = s1[j]; d2[j] = s2[j]; d3[j] = s3[j]; }
  }
#endif
  __syncthreads();

  const float scale = 0.125f;  // 64^-0.5

  // S = scale * Ql(256x64) * Qr^T(64x256): 256 16x16 tiles / 8 waves
  for (int t = wave; t < 256; t += 8) {
    int mt = t >> 4, ntile = t & 15;
    v8f acc = {};
#pragma unroll
    for (int kt = 0; kt < 2; ++kt) {
      v16h a  = frag_a_rm(sQl, 64, mt * 16, kt * 32, lane);
      v16h bb = frag_b_tr(sQr, 64, kt * 32, ntile * 16, lane);
      acc = wmma_f16(a, bb, acc);
    }
#pragma unroll
    for (int v = 0; v < 8; ++v) {
      int m = mt * 16 + hl * 8 + v, n = ntile * 16 + (lane & 15);
      sS[m * SLD + n] = (half_t)(acc[v] * scale);
    }
  }
  __syncthreads();

  // row max (thread = row w), then global max
  float rmax = -3.0e38f;
  {
    const half_t* row = sS + tid * SLD;
    for (int v = 0; v < 256; ++v) rmax = fmaxf(rmax, (float)row[v]);
  }
  sred[tid] = rmax;
  __syncthreads();
  for (int s = 128; s > 0; s >>= 1) {
    if (tid < s) sred[tid] = fmaxf(sred[tid], sred[tid + s]);
    __syncthreads();
  }
  float gmax = sred[0];

  // E1 = exp(S - rowmax) in place; rsum = row sums; rowscale = exp(rowmax - gmax)
  {
    half_t* row = sS + tid * SLD;
    float rs = 0.f;
    for (int v = 0; v < 256; ++v) {
      float e = __expf((float)row[v] - rmax);
      row[v] = (half_t)e;
      rs += e;
    }
    srsum[tid] = rs;
    srscl[tid] = __expf(rmax - gmax);
  }
  __syncthreads();

  // Z[v] = sum_w E1[w][v]*rowscale[w] (column softmax denominator, rescaled);
  // also scale V_l rows by rowscale[w] for the column-softmax GEMM
  {
    float z = 0.f;
    for (int w = 0; w < 256; ++w) z += (float)sS[w * SLD + tid] * srscl[w];
    sZ[tid] = z;
    float rs = srscl[tid];
    half_t* vrow = sVl + tid * 64;
    for (int c = 0; c < 64; ++c) vrow[c] = (half_t)((float)vrow[c] * rs);
  }
  __syncthreads();

  // F_r2l[w][c] = (E1 * Vr)[w][c] / rsum[w]
  for (int t = wave; t < 64; t += 8) {
    int mt = t >> 2, ntile = t & 3;
    v8f acc = {};
#pragma unroll
    for (int kt = 0; kt < 8; ++kt) {
      v16h a  = frag_a_rm(sS, SLD, mt * 16, kt * 32, lane);
      v16h bb = frag_b_rm(sVr, 64, kt * 32, ntile * 16, lane);
      acc = wmma_f16(a, bb, acc);
    }
#pragma unroll
    for (int v = 0; v < 8; ++v) {
      int m = mt * 16 + hl * 8 + v, n = ntile * 16 + (lane & 15);
      Fr2l[rowbase + (size_t)m * C_ + n] = (half_t)(acc[v] / srsum[m]);
    }
  }

  // F_l2r[v][c] = (E1^T * Vl_scaled)[v][c] / Z[v]   (transposed A reads; padded stride)
  for (int t = wave; t < 64; t += 8) {
    int mt = t >> 2, ntile = t & 3;
    v8f acc = {};
#pragma unroll
    for (int kt = 0; kt < 8; ++kt) {
      v16h a  = frag_a_tr(sS, SLD, mt * 16, kt * 32, lane);
      v16h bb = frag_b_rm(sVl, 64, kt * 32, ntile * 16, lane);
      acc = wmma_f16(a, bb, acc);
    }
#pragma unroll
    for (int v = 0; v < 8; ++v) {
      int m = mt * 16 + hl * 8 + v, n = ntile * 16 + (lane & 15);
      Fl2r[rowbase + (size_t)m * C_ + n] = (half_t)(acc[v] / sZ[m]);
    }
  }
}

// ---- Kernel 4: pattn1 = lp3*F_r2l + rp3*F_l2r + biases (accumulated WMMA GEMM), f32 NCHW ----
__global__ void __launch_bounds__(256) k_pattn(
    const half_t* F1, const half_t* F2,
    const float* w1, const float* b1, const float* w2, const float* b2,
    float* pattn) {
  extern __shared__ char smem[];
  half_t* sF1 = (half_t*)smem;        // 128 x 64 channels-last
  half_t* sF2 = sF1 + 128 * 64;
  half_t* sW1 = sF2 + 128 * 64;       // 64 x 64
  half_t* sW2 = sW1 + 64 * 64;
  int tid = threadIdx.x, lane = tid & 31, wave = tid >> 5, hl = lane >> 4;
  size_t n0 = (size_t)blockIdx.x * 128;
#if defined(HAVE_ASYNC)
  {
    const char* gA = (const char*)(F1 + n0 * 64);
    const char* gB = (const char*)(F2 + n0 * 64);
    char* lA = (char*)sF1;
    char* lB = (char*)sF2;
    for (int j = tid; j < 1024; j += 256) {
      __builtin_amdgcn_global_load_async_to_lds_b128(
          (gb128_p)(gA + j * 16), (lb128_p)(lA + j * 16), 0, 0);
      __builtin_amdgcn_global_load_async_to_lds_b128(
          (gb128_p)(gB + j * 16), (lb128_p)(lB + j * 16), 0, 0);
    }
    __builtin_amdgcn_s_wait_asynccnt(0);
  }
#else
  {
    const uint4* a = (const uint4*)(F1 + n0 * 64);
    const uint4* b = (const uint4*)(F2 + n0 * 64);
    uint4* da = (uint4*)sF1; uint4* db = (uint4*)sF2;
    for (int j = tid; j < 1024; j += 256) { da[j] = a[j]; db[j] = b[j]; }
  }
#endif
  for (int i = tid; i < 4096; i += 256) { sW1[i] = (half_t)w1[i]; sW2[i] = (half_t)w2[i]; }
  __syncthreads();

  int nt = wave * 16;
  for (int mt = 0; mt < 4; ++mt) {
    v8f acc = {};
#pragma unroll
    for (int kt = 0; kt < 2; ++kt) {
      v16h a  = frag_a_rm(sW1, 64, mt * 16, kt * 32, lane);
      v16h bb = frag_b_tr(sF1, 64, kt * 32, nt, lane);
      acc = wmma_f16(a, bb, acc);
    }
#pragma unroll
    for (int kt = 0; kt < 2; ++kt) {
      v16h a  = frag_a_rm(sW2, 64, mt * 16, kt * 32, lane);
      v16h bb = frag_b_tr(sF2, 64, kt * 32, nt, lane);
      acc = wmma_f16(a, bb, acc);
    }
    size_t n = n0 + nt + (lane & 15);
    int bb_ = (int)(n / HW_);
    size_t hw = n % HW_;
#pragma unroll
    for (int v = 0; v < 8; ++v) {
      int o = mt * 16 + hl * 8 + v;
      pattn[((size_t)(bb_ * C_ + o)) * HW_ + hw] = acc[v] + b1[o] + b2[o];
    }
  }
}

// ---- Kernel 5: pixel attention (7x7 grouped conv, reflect pad) + double sigmoid + blend ----
__device__ __forceinline__ int reflect_idx(int i, int n) {
  if (i < 0) i = -i;
  if (i >= n) i = 2 * n - 2 - i;
  return i;
}

__global__ void __launch_bounds__(256) k_pa_blend(
    const float* xl, const float* xr, const float* pattn,
    const float* paw, const float* pab, float* out) {
  __shared__ float sI[14][40];
  __shared__ float sP[14][40];
  __shared__ float sW[98];
  int tid = threadIdx.x;
  int bc = blockIdx.y; int c = bc & 63; int b = bc >> 6;
  int th = (blockIdx.x >> 3) * 8;     // tile origin in h (32 tiles)
  int tw = (blockIdx.x & 7) * 32;     // tile origin in w (8 tiles)
  const float* pxl = xl + (size_t)(b * C_ + c) * HW_;
  const float* pxr = xr + (size_t)(b * C_ + c) * HW_;
  const float* pp  = pattn + (size_t)(b * C_ + c) * HW_;

  if (tid < 98) sW[tid] = paw[c * 98 + tid];
  for (int i = tid; i < 14 * 38; i += 256) {
    int iy = i / 38, ix = i % 38;
    int yy = reflect_idx(th - 3 + iy, H_);
    int xx = reflect_idx(tw - 3 + ix, W_);
    sI[iy][ix] = pxl[yy * W_ + xx] + pxr[yy * W_ + xx];
    sP[iy][ix] = pp[yy * W_ + xx];
  }
  __syncthreads();

  int ty = tid >> 5, tx = tid & 31;
  float acc = pab[c];
#pragma unroll
  for (int i = 0; i < 7; ++i)
#pragma unroll
    for (int j = 0; j < 7; ++j)
      acc += sW[i * 7 + j] * sI[ty + i][tx + j] + sW[49 + i * 7 + j] * sP[ty + i][tx + j];

  float a1 = 1.f / (1.f + __expf(-acc));
  float a  = 1.f / (1.f + __expf(-a1));   // reference applies sigmoid twice
  int off = (th + ty) * W_ + (tw + tx);
  float vxl = pxl[off], vxr = pxr[off];
  out[(size_t)(b * C_ + c) * HW_ + off] = a * vxl + (1.f - a) * vxr;
}

extern "C" void kernel_launch(void* const* d_in, const int* in_sizes, int n_in,
                              void* d_out, int out_size, void* d_ws, size_t ws_size,
                              hipStream_t stream) {
  (void)in_sizes; (void)n_in; (void)out_size; (void)ws_size;
  const float* x_l    = (const float*)d_in[0];
  const float* x_r    = (const float*)d_in[1];
  const float* lp1_w1 = (const float*)d_in[2];
  const float* lp1_b1 = (const float*)d_in[3];
  const float* lp1_w2 = (const float*)d_in[4];
  const float* lp1_b2 = (const float*)d_in[5];
  const float* rp1_w1 = (const float*)d_in[6];
  const float* rp1_b1 = (const float*)d_in[7];
  const float* rp1_w2 = (const float*)d_in[8];
  const float* rp1_b2 = (const float*)d_in[9];
  const float* lp2_w1 = (const float*)d_in[10];
  const float* lp2_b1 = (const float*)d_in[11];
  const float* lp2_w2 = (const float*)d_in[12];
  const float* lp2_b2 = (const float*)d_in[13];
  const float* rp2_w1 = (const float*)d_in[14];
  const float* rp2_b1 = (const float*)d_in[15];
  const float* rp2_w2 = (const float*)d_in[16];
  const float* rp2_b2 = (const float*)d_in[17];
  const float* lp3_w  = (const float*)d_in[18];
  const float* lp3_b  = (const float*)d_in[19];
  const float* rp3_w  = (const float*)d_in[20];
  const float* rp3_b  = (const float*)d_in[21];
  const float* pa_w   = (const float*)d_in[22];
  const float* pa_b   = (const float*)d_in[23];

  char* ws = (char*)d_ws;
  const size_t NE = (size_t)B_ * C_ * H_ * W_;      // 8388608 elements
  half_t* t0 = (half_t*)(ws);
  half_t* t1 = (half_t*)(ws + NE * 2);
  half_t* t2 = (half_t*)(ws + NE * 4);
  half_t* t3 = (half_t*)(ws + NE * 6);
  half_t* Ql = (half_t*)(ws + NE * 8);
  half_t* Qr = (half_t*)(ws + NE * 10);
  half_t* Vl = (half_t*)(ws + NE * 12);
  half_t* Vr = (half_t*)(ws + NE * 14);
  half_t* F1 = t0;                       // F_r2l reuses t0 (dead after dw3x3)
  half_t* F2 = t1;                       // F_l2r reuses t1
  float*  pattn = (float*)(ws + NE * 4); // f32, reuses t2+t3 region (NE*4 bytes)

  size_t lds1 = (size_t)(64 * 128 + 64 * 64) * sizeof(half_t);
  k_conv1x1_qv<<<dim3(1024, 4), 256, lds1, stream>>>(
      x_l, x_r, lp1_w1, lp1_b1, rp1_w1, rp1_b1, lp2_w1, lp2_b1, rp2_w1, rp2_b1,
      t0, t1, t2, t3);

  k_dw3x3<<<dim3(32768, 4), 256, 0, stream>>>(
      t0, t1, t2, t3, lp1_w2, lp1_b2, rp1_w2, rp1_b2, lp2_w2, lp2_b2, rp2_w2, rp2_b2,
      Ql, Qr, Vl, Vr);

  size_t lds3 = (size_t)(4 * 256 * 64 + 256 * SLD) * sizeof(half_t) + 4 * 256 * sizeof(float);
  k_attn<<<dim3(B_ * H_), 256, lds3, stream>>>(Ql, Qr, Vl, Vr, F1, F2);

  size_t lds4 = (size_t)(2 * 128 * 64 + 2 * 64 * 64) * sizeof(half_t);
  k_pattn<<<dim3(1024), 256, lds4, stream>>>(F1, F2, lp3_w, lp3_b, rp3_w, rp3_b, pattn);

  k_pa_blend<<<dim3(256, B_ * C_), 256, 0, stream>>>(x_l, x_r, pattn, pa_w, pa_b, (float*)d_out);
}
